// ConformerLayer_63677185131392
// MI455X (gfx1250) — compile-verified, matrix operand order checked by reading, not run on hardware
//
#include <hip/hip_runtime.h>
#include <hip/hip_bf16.h>

// ---------------------------------------------------------------------------
// Conformer layer for MI455X (gfx1250): bf16 WMMA GEMMs, async global->LDS
// tile staging, fused epilogues.
// T=2048, B=16, D=256, H=4, DH=64, F=1024, K(conv)=31, MRP=2. Tokens M=T*B=32768.
// ---------------------------------------------------------------------------

typedef __attribute__((ext_vector_type(16))) __bf16 v16bf;
typedef __attribute__((ext_vector_type(8)))  float  v8f;
typedef __attribute__((ext_vector_type(4)))  int    v4i;

#define TOKENS 32768
#define DMODEL 256
#define FDIM   1024
#define NHEAD  4
#define DHEAD  64
#define TLEN   2048
#define BSEQ   16

__device__ __forceinline__ float siluf(float x) { return x / (1.f + __expf(-x)); }

// ---- CDNA5 async global->LDS copy (ASYNCcnt path), with safe fallback ------
#if __has_builtin(__builtin_amdgcn_global_load_async_to_lds_b128)
#define HAVE_ASYNC_LDS 1
#endif

__device__ __forceinline__ void async_b128(const __bf16* g, __bf16* l) {
#ifdef HAVE_ASYNC_LDS
    __builtin_amdgcn_global_load_async_to_lds_b128(
        (__attribute__((address_space(1))) v4i*)g,
        (__attribute__((address_space(3))) v4i*)l, 0, 0);
#else
    *(uint4*)l = *(const uint4*)g;
#endif
}

__device__ __forceinline__ void wait_async_lds() {
#ifdef HAVE_ASYNC_LDS
#if __has_builtin(__builtin_amdgcn_s_wait_asynccnt)
    __builtin_amdgcn_s_wait_asynccnt(0);
#else
    asm volatile("s_wait_asynccnt 0x0" ::: "memory");
#endif
#endif
}

// ---------------------------------------------------------------------------
// fp32 -> bf16 conversion (weights)
// ---------------------------------------------------------------------------
__global__ void cvt_f32_bf16(const float* __restrict__ s, __bf16* __restrict__ d, int n) {
    int i = blockIdx.x * 256 + threadIdx.x;
    if (i < n) d[i] = (__bf16)s[i];
}

// ---------------------------------------------------------------------------
// LayerNorm over D=256 (one block = one row), optional bf16 output
// ---------------------------------------------------------------------------
template<bool OUTB>
__global__ __launch_bounds__(256) void ln256(const float* __restrict__ x,
                                             const float* __restrict__ g,
                                             const float* __restrict__ bb,
                                             void* __restrict__ out) {
    __shared__ float red[256];
    size_t row = blockIdx.x;
    int c = threadIdx.x;
    float v = x[row * DMODEL + c];
    red[c] = v; __syncthreads();
    for (int s = 128; s > 0; s >>= 1) { if (c < s) red[c] += red[c + s]; __syncthreads(); }
    float mean = red[0] * (1.f / 256.f);
    __syncthreads();
    float dv = v - mean;
    red[c] = dv * dv; __syncthreads();
    for (int s = 128; s > 0; s >>= 1) { if (c < s) red[c] += red[c + s]; __syncthreads(); }
    float var = red[0] * (1.f / 256.f);
    float y = dv * rsqrtf(var + 1e-5f) * g[c] + bb[c];
    if (OUTB) ((__bf16*)out)[row * DMODEL + c] = (__bf16)y;
    else      ((float*)out)[row * DMODEL + c] = y;
}

// ---------------------------------------------------------------------------
// WMMA GEMM: out[M,N] = act( A[M,K](bf16) @ W[N,K]^T(bf16) + bias ) * alpha + res
// Block tile 128x128, 8 waves (4 in M x 2 in N); each wave 32x64 = 2x4 WMMA
// accumulators (v_wmma_f32_16x16x32_bf16). A/B tiles staged to LDS with
// GLOBAL_LOAD_ASYNC_TO_LDS_B128 (16B-contiguous transfers; B tile stored
// n-major so its global rows are contiguous in K).
// ACT: 0=none, 1=silu. OUTB: bf16 output. RES: add fp32 residual.
// ---------------------------------------------------------------------------
template<int ACT, bool OUTB, bool RES>
__global__ __launch_bounds__(256) void gemm_wmma(const __bf16* __restrict__ A,
                                                 const __bf16* __restrict__ W,
                                                 const float* __restrict__ bias,
                                                 const float* __restrict__ res,
                                                 void* __restrict__ outp,
                                                 int M, int N, int K, float alpha) {
    __shared__ __bf16 As[128][40];   // rows = m, 80B row stride (16B aligned, bank-spread)
    __shared__ __bf16 Bs[128][40];   // rows = n, cols = k (n-major!)
    const int tid  = threadIdx.x;
    const int wave = tid >> 5;
    const int lane = tid & 31;
    const int hf   = lane >> 4;   // 0/1: which 16-lane half
    const int l16  = lane & 15;
    const int wm   = wave >> 1;   // 0..3 -> 32-row strip
    const int wn   = wave & 1;    // 0..1 -> 64-col strip
    const int m0   = blockIdx.x * 128;
    const int n0   = blockIdx.y * 128;

    v8f acc[2][4] = {};

    for (int k0 = 0; k0 < K; k0 += 32) {
        // A tile: 128 rows x 32 halfs = 512 x 16B transfers, 2 per thread
        #pragma unroll
        for (int i = 0; i < 2; i++) {
            int idx = tid * 2 + i;
            int r = idx >> 2, seg = idx & 3;
            async_b128(&A[(size_t)(m0 + r) * K + k0 + seg * 8], &As[r][seg * 8]);
        }
        // B tile (n-major): 128 n-rows x 32 halfs, contiguous in K in global
        #pragma unroll
        for (int i = 0; i < 2; i++) {
            int idx = tid * 2 + i;
            int n = idx >> 2, seg = idx & 3;
            async_b128(&W[(size_t)(n0 + n) * K + k0 + seg * 8], &Bs[n][seg * 8]);
        }
        wait_async_lds();
        __syncthreads();

        v16bf af[2], bw[4];
        #pragma unroll
        for (int i = 0; i < 2; i++) {
            int m = wm * 32 + i * 16 + l16;
            #pragma unroll
            for (int vv = 0; vv < 8; vv++) {
                int kk = (vv >> 2) * 16 + hf * 8 + (vv & 3) * 2;  // ISA 16-bit A 16x32 layout
                af[i][2 * vv]     = As[m][kk];
                af[i][2 * vv + 1] = As[m][kk + 1];
            }
        }
        #pragma unroll
        for (int j = 0; j < 4; j++) {
            int n = wn * 64 + j * 16 + l16;
            #pragma unroll
            for (int vv = 0; vv < 8; vv++) {
                int kk = (vv >> 2) * 16 + hf * 8 + (vv & 3) * 2;  // B 32x16 layout (transpose-sym)
                bw[j][2 * vv]     = Bs[n][kk];
                bw[j][2 * vv + 1] = Bs[n][kk + 1];
            }
        }
        #pragma unroll
        for (int i = 0; i < 2; i++)
            #pragma unroll
            for (int j = 0; j < 4; j++)
                acc[i][j] = __builtin_amdgcn_wmma_f32_16x16x32_bf16(
                    false, af[i], false, bw[j], (short)0, acc[i][j], false, false);
        __syncthreads();
    }

    #pragma unroll
    for (int i = 0; i < 2; i++) {
        #pragma unroll
        for (int j = 0; j < 4; j++) {
            int n = n0 + wn * 64 + j * 16 + l16;
            float bv = bias[n];
            #pragma unroll
            for (int r = 0; r < 8; r++) {
                int m = m0 + wm * 32 + i * 16 + hf * 8 + r;   // C layout: M = r + 8*half
                float val = acc[i][j][r] + bv;
                if (ACT == 1) val = siluf(val);
                val *= alpha;
                if (RES) val += res[(size_t)m * N + n];
                if (OUTB) ((__bf16*)outp)[(size_t)m * N + n] = (__bf16)val;
                else      ((float*)outp)[(size_t)m * N + n] = val;
            }
        }
    }
}

// ---------------------------------------------------------------------------
// Attention: one wave per (t, h). Sequence axis = B = 16, head dim = 64.
// scores(16x16) = qh(16x64)@kh^T (2 WMMAs) + rel_k bias; softmax over k;
// ctx = p@vh (4 WMMAs, K padded 16->32) + rel_v term.
// ---------------------------------------------------------------------------
__global__ __launch_bounds__(32) void attn_kernel(const __bf16* __restrict__ q,
                                                  const __bf16* __restrict__ k,
                                                  const __bf16* __restrict__ v,
                                                  const float* __restrict__ rel_k,
                                                  const float* __restrict__ rel_v,
                                                  __bf16* __restrict__ ctx) {
    const int t = blockIdx.x;
    const int h = blockIdx.y;
    const int lane = threadIdx.x;
    const int hf  = lane >> 4;
    const int l16 = lane & 15;

    __shared__ float  qs[16][64];
    __shared__ float  ps[16][16];
    __shared__ __bf16 pb[16][32];

    for (int idx = lane; idx < 16 * 64; idx += 32) {
        int b = idx >> 6, d = idx & 63;
        qs[b][d] = (float)q[((size_t)t * BSEQ + b) * DMODEL + h * DHEAD + d];
    }
    __syncthreads();

    // scores = qh @ kh^T (contract over DHEAD=64 -> two K=32 WMMA steps)
    v8f sacc = {};
    for (int kk0 = 0; kk0 < 64; kk0 += 32) {
        v16bf af = {}, bw = {};
        #pragma unroll
        for (int vv = 0; vv < 8; vv++) {
            int kk = (vv >> 2) * 16 + hf * 8 + (vv & 3) * 2;
            af[2 * vv]     = (__bf16)qs[l16][kk0 + kk];
            af[2 * vv + 1] = (__bf16)qs[l16][kk0 + kk + 1];
            size_t kb = ((size_t)t * BSEQ + l16) * DMODEL + h * DHEAD + kk0 + kk;
            bw[2 * vv]     = k[kb];
            bw[2 * vv + 1] = k[kb + 1];
        }
        sacc = __builtin_amdgcn_wmma_f32_16x16x32_bf16(false, af, false, bw, (short)0, sacc, false, false);
    }

    // add relative-position bias (table idx = clip(k-q,-2,2)+2), scale 1/sqrt(64)
    #pragma unroll
    for (int r = 0; r < 8; r++) {
        int qi = r + hf * 8;
        int ki = l16;
        int rr = ki - qi; rr = rr < -2 ? -2 : (rr > 2 ? 2 : rr);
        const float* rk = rel_k + (rr + 2) * DHEAD;
        float a2 = 0.f;
        for (int d = 0; d < DHEAD; d++) a2 += qs[qi][d] * rk[d];
        ps[qi][ki] = (sacc[r] + a2) * 0.125f;
    }
    __syncthreads();

    // softmax over k: 16 lanes handle 16 rows
    if (lane < 16) {
        float mx = -1e30f;
        for (int kk = 0; kk < 16; kk++) mx = fmaxf(mx, ps[lane][kk]);
        float s = 0.f;
        for (int kk = 0; kk < 16; kk++) { float e = __expf(ps[lane][kk] - mx); ps[lane][kk] = e; s += e; }
        float inv = 1.f / s;
        for (int kk = 0; kk < 16; kk++) { float p = ps[lane][kk] * inv; ps[lane][kk] = p; pb[lane][kk] = (__bf16)p; }
        for (int kk = 16; kk < 32; kk++) pb[lane][kk] = (__bf16)0.f;
    }
    __syncthreads();

    // ctx = p(16x32, zero-padded) @ vh(32x16 tiles), plus rel_v term
    v16bf pf = {};
    #pragma unroll
    for (int vv = 0; vv < 8; vv++) {
        int kk = (vv >> 2) * 16 + hf * 8 + (vv & 3) * 2;
        pf[2 * vv]     = pb[l16][kk];
        pf[2 * vv + 1] = pb[l16][kk + 1];
    }
    for (int dt = 0; dt < 4; dt++) {
        v16bf vf = {};
        #pragma unroll
        for (int vv = 0; vv < 8; vv++) {
            int kk = (vv >> 2) * 16 + hf * 8 + (vv & 3) * 2;
            if (kk < 16) {
                size_t vb = ((size_t)t * BSEQ + kk) * DMODEL + h * DHEAD + dt * 16 + l16;
                vf[2 * vv]     = v[vb];
                vf[2 * vv + 1] = v[vb + DMODEL];   // row kk+1
            } else {
                vf[2 * vv] = (__bf16)0.f; vf[2 * vv + 1] = (__bf16)0.f;
            }
        }
        v8f oacc = {};
        oacc = __builtin_amdgcn_wmma_f32_16x16x32_bf16(false, pf, false, vf, (short)0, oacc, false, false);
        #pragma unroll
        for (int r = 0; r < 8; r++) {
            int qi = r + hf * 8;
            int d = dt * 16 + l16;
            float w2 = 0.f;
            for (int kk = 0; kk < 16; kk++) {
                int rr = kk - qi; rr = rr < -2 ? -2 : (rr > 2 ? 2 : rr);
                w2 += ps[qi][kk] * rel_v[(rr + 2) * DHEAD + d];
            }
            ctx[((size_t)t * BSEQ + qi) * DMODEL + h * DHEAD + d] = (__bf16)(oacc[r] + w2);
        }
    }
}

// ---------------------------------------------------------------------------
// GLU: in (tokens,512) fp32 -> out (tokens,256) bf16, pair (c, c+256)
// ---------------------------------------------------------------------------
__global__ void glu_k(const float* __restrict__ in, __bf16* __restrict__ out) {
    size_t idx = (size_t)blockIdx.x * 256 + threadIdx.x;
    size_t row = idx >> 8;
    int c = (int)(idx & 255);
    float a = in[row * 512 + c];
    float g = in[row * 512 + 256 + c];
    out[idx] = (__bf16)(a * (1.f / (1.f + __expf(-g))));
}

// ---------------------------------------------------------------------------
// Depthwise conv over time T, K=31, pad 15. Layout (t,b,c), c fastest.
// ---------------------------------------------------------------------------
__global__ void dwconv_k(const __bf16* __restrict__ x, const float* __restrict__ w,
                         const float* __restrict__ bias, float* __restrict__ out) {
    size_t idx = (size_t)blockIdx.x * 256 + threadIdx.x;
    int c = (int)(idx & 255);
    int b = (int)((idx >> 8) & 15);
    int t = (int)(idx >> 12);
    float acc = bias[c];
    for (int kk = 0; kk < 31; kk++) {
        int tt = t + kk - 15;
        if (tt >= 0 && tt < TLEN)
            acc += w[c * 31 + kk] * (float)x[((size_t)tt * BSEQ + b) * DMODEL + c];
    }
    out[idx] = acc;
}

// ---------------------------------------------------------------------------
// GroupNorm(1 group) over (D,T) per batch b: stats then apply+silu (bf16 out)
// ---------------------------------------------------------------------------
__global__ void gn_zero(float* p) { if (threadIdx.x < 32) p[threadIdx.x] = 0.f; }

__global__ __launch_bounds__(256) void gn_stats(const float* __restrict__ x, float* __restrict__ stats) {
    __shared__ float s1[256], s2[256];
    size_t idx = (size_t)blockIdx.x * 256 + threadIdx.x;
    int b = (int)((idx >> 8) & 15);
    float v = x[idx];
    s1[threadIdx.x] = v; s2[threadIdx.x] = v * v; __syncthreads();
    for (int s = 128; s > 0; s >>= 1) {
        if ((int)threadIdx.x < s) { s1[threadIdx.x] += s1[threadIdx.x + s]; s2[threadIdx.x] += s2[threadIdx.x + s]; }
        __syncthreads();
    }
    if (threadIdx.x == 0) { atomicAdd(&stats[b], s1[0]); atomicAdd(&stats[16 + b], s2[0]); }
}

__global__ void gn_apply(const float* __restrict__ x, const float* __restrict__ stats,
                         const float* __restrict__ g, const float* __restrict__ bta,
                         __bf16* __restrict__ out) {
    size_t idx = (size_t)blockIdx.x * 256 + threadIdx.x;
    int c = (int)(idx & 255);
    int b = (int)((idx >> 8) & 15);
    const float invn = 1.f / (2048.f * 256.f);
    float mean = stats[b] * invn;
    float var  = stats[16 + b] * invn - mean * mean;
    float y = (x[idx] - mean) * rsqrtf(var + 1e-5f) * g[c] + bta[c];
    out[idx] = (__bf16)siluf(y);
}

// ---------------------------------------------------------------------------
// Driver
// ---------------------------------------------------------------------------
extern "C" void kernel_launch(void* const* d_in, const int* in_sizes, int n_in,
                              void* d_out, int out_size, void* d_ws, size_t ws_size,
                              hipStream_t stream) {
    (void)in_sizes; (void)n_in; (void)out_size; (void)ws_size;

    const float* input     = (const float*)d_in[0];
    const float* ffn1_ln_g = (const float*)d_in[3];
    const float* ffn1_ln_b = (const float*)d_in[4];
    const float* ffn1_w1   = (const float*)d_in[5];
    const float* ffn1_b1   = (const float*)d_in[6];
    const float* ffn1_w2   = (const float*)d_in[7];
    const float* ffn1_b2   = (const float*)d_in[8];
    const float* attn_ln_g = (const float*)d_in[9];
    const float* attn_ln_b = (const float*)d_in[10];
    const float* wq = (const float*)d_in[11]; const float* bq = (const float*)d_in[12];
    const float* wk = (const float*)d_in[13]; const float* bk = (const float*)d_in[14];
    const float* wv = (const float*)d_in[15]; const float* bv = (const float*)d_in[16];
    const float* wo = (const float*)d_in[17]; const float* bo = (const float*)d_in[18];
    const float* rel_k = (const float*)d_in[19];
    const float* rel_v = (const float*)d_in[20];
    const float* conv_ln_g = (const float*)d_in[21];
    const float* conv_ln_b = (const float*)d_in[22];
    const float* pw1_w = (const float*)d_in[23]; const float* pw1_b = (const float*)d_in[24];
    const float* dw_w  = (const float*)d_in[25]; const float* dw_b  = (const float*)d_in[26];
    const float* gn_g  = (const float*)d_in[27]; const float* gn_b  = (const float*)d_in[28];
    const float* pw2_w = (const float*)d_in[29]; const float* pw2_b = (const float*)d_in[30];
    const float* ffn2_ln_g = (const float*)d_in[31];
    const float* ffn2_ln_b = (const float*)d_in[32];
    const float* ffn2_w1 = (const float*)d_in[33]; const float* ffn2_b1 = (const float*)d_in[34];
    const float* ffn2_w2 = (const float*)d_in[35]; const float* ffn2_b2 = (const float*)d_in[36];
    const float* final_ln_g = (const float*)d_in[37];
    const float* final_ln_b = (const float*)d_in[38];

    char* ws = (char*)d_ws;
    float*  X     = (float*) (ws + 0);           // 32768x256 f32  (33.5 MB) residual stream
    __bf16* BF_IN = (__bf16*)(ws + 33554432);    // 32768x256 bf16 (16.8 MB) LN output / attn ctx
    __bf16* BF_H  = (__bf16*)(ws + 50331648);    // 32768x1024 bf16 (67 MB) FFN hidden / QKV / GLU
    float*  F32_T = (float*) (ws + 117440512);   // 32768x512 f32  (67 MB) pw1 out / dwconv out
    __bf16* WB    = (__bf16*)(ws + 184549376);   // bf16 weight copies (~3 MB)
    float*  STATS = (float*) (ws + 184549376 + 3014656);

    __bf16* w1b   = WB;
    __bf16* w2b   = WB + 262144;
    __bf16* wqb   = WB + 524288;
    __bf16* wkb   = WB + 589824;
    __bf16* wvb   = WB + 655360;
    __bf16* wob   = WB + 720896;
    __bf16* pw1wb = WB + 786432;
    __bf16* pw2wb = WB + 917504;
    __bf16* f2w1b = WB + 983040;
    __bf16* f2w2b = WB + 1245184;

    __bf16* Qb = BF_H;
    __bf16* Kb = BF_H + 8388608;
    __bf16* Vb = BF_H + 16777216;

    auto cvt = [&](const float* s, __bf16* d, int n) {
        cvt_f32_bf16<<<(n + 255) / 256, 256, 0, stream>>>(s, d, n);
    };
    cvt(ffn1_w1, w1b, 262144);
    cvt(ffn1_w2, w2b, 262144);
    cvt(wq, wqb, 65536);
    cvt(wk, wkb, 65536);
    cvt(wv, wvb, 65536);
    cvt(wo, wob, 65536);
    cvt(pw1_w, pw1wb, 131072);
    cvt(pw2_w, pw2wb, 65536);
    cvt(ffn2_w1, f2w1b, 262144);
    cvt(ffn2_w2, f2w2b, 262144);

    const dim3 gemmB(256);
    const dim3 gF1(TOKENS / 128, FDIM / 128);    // (256,8)
    const dim3 gF2(TOKENS / 128, DMODEL / 128);  // (256,2)
    const dim3 gPW1(TOKENS / 128, 512 / 128);    // (256,4)
    const int  ROWS = TOKENS;
    const int  ELT  = TOKENS * DMODEL / 256;     // 32768 blocks of 256

    // ---- FFN1: x = 0.5*FFN(ln(input)) + input ----
    ln256<true><<<ROWS, 256, 0, stream>>>(input, ffn1_ln_g, ffn1_ln_b, BF_IN);
    gemm_wmma<1, true,  false><<<gF1, gemmB, 0, stream>>>(BF_IN, w1b, ffn1_b1, nullptr, BF_H,
                                                          TOKENS, FDIM, DMODEL, 1.0f);
    gemm_wmma<0, false, true ><<<gF2, gemmB, 0, stream>>>(BF_H, w2b, ffn1_b2, input, X,
                                                          TOKENS, DMODEL, FDIM, 0.5f);

    // ---- Attention (over B axis) ----
    ln256<true><<<ROWS, 256, 0, stream>>>(X, attn_ln_g, attn_ln_b, BF_IN);
    gemm_wmma<0, true, false><<<gF2, gemmB, 0, stream>>>(BF_IN, wqb, bq, nullptr, Qb,
                                                         TOKENS, DMODEL, DMODEL, 1.0f);
    gemm_wmma<0, true, false><<<gF2, gemmB, 0, stream>>>(BF_IN, wkb, bk, nullptr, Kb,
                                                         TOKENS, DMODEL, DMODEL, 1.0f);
    gemm_wmma<0, true, false><<<gF2, gemmB, 0, stream>>>(BF_IN, wvb, bv, nullptr, Vb,
                                                         TOKENS, DMODEL, DMODEL, 1.0f);
    attn_kernel<<<dim3(TLEN, NHEAD), 32, 0, stream>>>(Qb, Kb, Vb, rel_k, rel_v, BF_IN);
    gemm_wmma<0, false, true><<<gF2, gemmB, 0, stream>>>(BF_IN, wob, bo, X, X,
                                                         TOKENS, DMODEL, DMODEL, 1.0f);

    // ---- Conv module: x = conv(x) + x ----
    ln256<true><<<ROWS, 256, 0, stream>>>(X, conv_ln_g, conv_ln_b, BF_IN);
    gemm_wmma<0, false, false><<<gPW1, gemmB, 0, stream>>>(BF_IN, pw1wb, pw1_b, nullptr, F32_T,
                                                           TOKENS, 512, DMODEL, 1.0f);
    glu_k<<<ELT, 256, 0, stream>>>(F32_T, BF_H);
    dwconv_k<<<ELT, 256, 0, stream>>>(BF_H, dw_w, dw_b, F32_T);
    gn_zero<<<1, 32, 0, stream>>>(STATS);
    gn_stats<<<ELT, 256, 0, stream>>>(F32_T, STATS);
    gn_apply<<<ELT, 256, 0, stream>>>(F32_T, STATS, gn_g, gn_b, BF_IN);
    gemm_wmma<0, false, true><<<gF2, gemmB, 0, stream>>>(BF_IN, pw2wb, pw2_b, X, X,
                                                         TOKENS, DMODEL, DMODEL, 1.0f);

    // ---- FFN2: x = 0.5*FFN(ln(x)) + x ----
    ln256<true><<<ROWS, 256, 0, stream>>>(X, ffn2_ln_g, ffn2_ln_b, BF_IN);
    gemm_wmma<1, true,  false><<<gF1, gemmB, 0, stream>>>(BF_IN, f2w1b, ffn2_b1, nullptr, BF_H,
                                                          TOKENS, FDIM, DMODEL, 1.0f);
    gemm_wmma<0, false, true ><<<gF2, gemmB, 0, stream>>>(BF_H, f2w2b, ffn2_b2, X, X,
                                                          TOKENS, DMODEL, FDIM, 0.5f);

    // ---- Final LN -> d_out (fp32) ----
    ln256<false><<<ROWS, 256, 0, stream>>>(X, final_ln_g, final_ln_b, (float*)d_out);
}
